// LSTMSeq2seq_17566416241424
// MI455X (gfx1250) — compile-verified, matrix operand order checked by reading, not run on hardware
//
#include <hip/hip_runtime.h>
#include <math.h>

// Problem constants (match reference)
#define B_   128
#define L_   64
#define T_   48
#define E_   512
#define H_   512
#define G4H  2048          // 4*H
#define VT   32000
#define NEGBIG (-1e9f)

typedef __attribute__((ext_vector_type(16))) _Float16 v16h;
typedef __attribute__((ext_vector_type(8)))  float    v8f;

// ---------------------------------------------------------------------------
// Fragment load: per CDNA5 ISA 16-bit A/B layout, each lane holds two
// contiguous 8-half (16B) chunks: K = kbase+[0..7] and kbase+16+[0..7],
// where kbase = (lane<16 ? 0 : 8). Identical pattern for A (row=M) and
// B (row=N of the (N,K) weight matrix).
// ---------------------------------------------------------------------------
__device__ __forceinline__ v16h ld_frag(const _Float16* p) {
  v16h f;
  float4* d = reinterpret_cast<float4*>(&f);
  d[0] = *reinterpret_cast<const float4*>(p);
  d[1] = *reinterpret_cast<const float4*>(p + 16);
  return f;
}

// ---------------------------------------------------------------------------
// GEMM: C[M,N] = A[M,K] * W[N,K]^T (+ addend) (+ bias), f16 in / f32 acc.
// Block = 128 threads = 4 waves; each wave computes a 16x64 tile (A fragment
// reused across 4 WMMAs). Requires M%64==0, N%64==0, K%32==0 (all shapes in
// this model satisfy that: M in {128, 8192}, N in {512, 2048, 32000},
// K in {512, 1024}).
// ---------------------------------------------------------------------------
__global__ __launch_bounds__(128) void gemm_f16_wmma(
    const _Float16* __restrict__ A, int lda,
    const _Float16* __restrict__ W, int ldb,
    const float* __restrict__ addend,
    const float* __restrict__ bias,
    float* __restrict__ C, _Float16* __restrict__ C16,
    int ldc, int K)
{
  const int wave = threadIdx.x >> 5;
  const int lane = threadIdx.x & 31;
  const int hsel = lane >> 4;      // 0: K 0..7/16..23, 1: K 8..15/24..31
  const int r    = lane & 15;
  const int row0 = blockIdx.y * 64 + wave * 16;
  const int col0 = blockIdx.x * 64;

  const _Float16* aP  = A + (size_t)(row0 + r) * lda + (hsel << 3);
  const _Float16* bP0 = W + (size_t)(col0 +  0 + r) * ldb + (hsel << 3);
  const _Float16* bP1 = W + (size_t)(col0 + 16 + r) * ldb + (hsel << 3);
  const _Float16* bP2 = W + (size_t)(col0 + 32 + r) * ldb + (hsel << 3);
  const _Float16* bP3 = W + (size_t)(col0 + 48 + r) * ldb + (hsel << 3);

  v8f acc0 = {}, acc1 = {}, acc2 = {}, acc3 = {};
  for (int k = 0; k < K; k += 32) {
    v16h a  = ld_frag(aP  + k);
    v16h b0 = ld_frag(bP0 + k);
    v16h b1 = ld_frag(bP1 + k);
    v16h b2 = ld_frag(bP2 + k);
    v16h b3 = ld_frag(bP3 + k);
    acc0 = __builtin_amdgcn_wmma_f32_16x16x32_f16(false, a, false, b0, (short)0, acc0, false, false);
    acc1 = __builtin_amdgcn_wmma_f32_16x16x32_f16(false, a, false, b1, (short)0, acc1, false, false);
    acc2 = __builtin_amdgcn_wmma_f32_16x16x32_f16(false, a, false, b2, (short)0, acc2, false, false);
    acc3 = __builtin_amdgcn_wmma_f32_16x16x32_f16(false, a, false, b3, (short)0, acc3, false, false);
  }

  v8f accs[4] = {acc0, acc1, acc2, acc3};
#pragma unroll
  for (int jt = 0; jt < 4; ++jt) {
    const int cc = col0 + jt * 16 + r;
#pragma unroll
    for (int i = 0; i < 8; ++i) {
      // C/D layout: VGPR i -> row = row0 + i + 8*hsel, col = col0 + 16*jt + (lane&15)
      const int rr = row0 + (hsel << 3) + i;
      const size_t idx = (size_t)rr * ldc + cc;
      float v = accs[jt][i];
      if (addend) v += addend[idx];
      if (bias)   v += bias[cc];
      C[idx] = v;
      if (C16) C16[idx] = (_Float16)v;
    }
  }
}

// ---------------------------------------------------------------------------
// Elementwise / gather / reduction kernels
// ---------------------------------------------------------------------------
__global__ void f32_to_f16_k(const float* __restrict__ s, _Float16* __restrict__ d, int n) {
  int i = blockIdx.x * 256 + threadIdx.x;
  if (i < n) d[i] = (_Float16)s[i];
}

// x16[(l*B + b)*E + e] = src_emb[src_tokens[b,l], e]   (exact grid, no bounds check)
__global__ void gather_src_k(const int* __restrict__ tok, const float* __restrict__ emb,
                             _Float16* __restrict__ x16) {
  int i = blockIdx.x * 256 + threadIdx.x;      // < L*B*E = 4194304
  int e = i & (E_ - 1);
  int rbl = i >> 9;                            // l*B + b
  int l = rbl >> 7;
  int b = rbl & (B_ - 1);
  int tk = tok[b * L_ + l];
  x16[i] = (_Float16)emb[(size_t)tk * E_ + e];
}

__device__ __forceinline__ float sigm(float x) { return 1.f / (1.f + expf(-x)); }

// LSTM gate math. pre = (B,4H) preactivation (i|f|g|o), c/h f32 state.
// h16: masked new h in f16 (for next-step recurrence GEMM).
// nh16: UNMASKED new h in f16 (layer-1 input — reference feeds pre-mask nh1).
// outs: src_states (B,L,H), written as mask ? nh : 0.
__global__ void lstm_gate_k(const float* __restrict__ pre,
                            float* __restrict__ h, float* __restrict__ c,
                            _Float16* __restrict__ h16, _Float16* __restrict__ nh16,
                            float* __restrict__ outs, const int* __restrict__ lens, int l)
{
  int idx = blockIdx.x * 256 + threadIdx.x;    // < B*H = 65536
  int b = idx >> 9;
  int j = idx & (H_ - 1);
  const float* g = pre + (size_t)b * G4H;
  float i_ = sigm(g[j]);
  float f_ = sigm(g[H_ + j]);
  float gg = tanhf(g[2 * H_ + j]);
  float o_ = sigm(g[3 * H_ + j]);
  float nc = f_ * c[idx] + i_ * gg;
  float nh = o_ * tanhf(nc);
  bool m = lens ? (l < lens[b]) : true;
  float hn = m ? nh : h[idx];
  float cn = m ? nc : c[idx];
  h[idx] = hn; c[idx] = cn;
  if (h16)  h16[idx]  = (_Float16)hn;
  if (nh16) nh16[idx] = (_Float16)nh;
  if (outs) outs[((size_t)b * L_ + l) * H_ + j] = m ? nh : 0.f;
}

// cellcat16[b, 0:H] = c1[b], cellcat16[b, H:2H] = c2[b]
__global__ void cellcat_k(const float* __restrict__ c1, const float* __restrict__ c2,
                          _Float16* __restrict__ out) {
  int i = blockIdx.x * 256 + threadIdx.x;      // < B*2H
  int b = i >> 10;
  int j = i & 1023;
  float v = (j < H_) ? c1[b * H_ + j] : c2[b * H_ + (j - H_)];
  out[i] = (_Float16)v;
}

__global__ void tanh_k(const float* __restrict__ c, float* __restrict__ h,
                       _Float16* __restrict__ h16) {
  int i = blockIdx.x * 256 + threadIdx.x;      // < B*H
  float v = tanhf(c[i]);
  h[i] = v; h16[i] = (_Float16)v;
}

// decoder input concat: [trg_emb[tok_t]; av_prev] -> f16 (B, E+H)
__global__ void dec_input_k(const int* __restrict__ trg_tokens, const float* __restrict__ emb,
                            const float* __restrict__ av, _Float16* __restrict__ in16, int t) {
  int b = blockIdx.x, i = threadIdx.x;         // block 1024
  float v;
  if (i < E_) { int tk = trg_tokens[b * T_ + t]; v = emb[(size_t)tk * E_ + i]; }
  else        { v = av[b * H_ + (i - E_)]; }
  in16[(size_t)b * (E_ + H_) + i] = (_Float16)v;
}

// scores + masked softmax over L=64 src positions. grid=B, block=64.
__global__ void attn_sm_k(const float* __restrict__ hdec, const float* __restrict__ ss,
                          const int* __restrict__ lens, float* __restrict__ attn) {
  __shared__ float hs[H_];
  __shared__ float red[L_];
  int b = blockIdx.x, t = threadIdx.x;
  for (int k = t; k < H_; k += L_) hs[k] = hdec[b * H_ + k];
  __syncthreads();
  const float* row = ss + ((size_t)b * L_ + t) * H_;
  float s = 0.f;
  for (int k = 0; k < H_; ++k) s += hs[k] * row[k];
  if (t >= lens[b]) s = NEGBIG;
  red[t] = s; __syncthreads();
  for (int o = 32; o > 0; o >>= 1) { if (t < o) red[t] = fmaxf(red[t], red[t + o]); __syncthreads(); }
  float mx = red[0]; __syncthreads();
  float e = expf(s - mx);
  red[t] = e; __syncthreads();
  for (int o = 32; o > 0; o >>= 1) { if (t < o) red[t] += red[t + o]; __syncthreads(); }
  attn[b * L_ + t] = e / red[0];
}

// ctx[b,h] = sum_l attn*src_states; writes concat2 = [h_dec; ctx] in f16. grid=B, block=512.
__global__ void ctx_concat_k(const float* __restrict__ attn, const float* __restrict__ ss,
                             const float* __restrict__ hdec, _Float16* __restrict__ cat) {
  __shared__ float a[L_];
  int b = blockIdx.x, j = threadIdx.x;
  if (j < L_) a[j] = attn[b * L_ + j];
  __syncthreads();
  float s = 0.f;
  for (int l = 0; l < L_; ++l) s += a[l] * ss[((size_t)b * L_ + l) * H_ + j];
  cat[(size_t)b * (2 * H_) + j]      = (_Float16)hdec[b * H_ + j];
  cat[(size_t)b * (2 * H_) + H_ + j] = (_Float16)s;
}

// per-step NLL: out += logp[tgt] if (t+1) < trg_lens[b]. grid=B, block=256.
__global__ void nll_step_k(const float* __restrict__ logits, const int* __restrict__ trg_tokens,
                           const int* __restrict__ trg_lens, float* __restrict__ out, int t) {
  __shared__ float red[256];
  const int b = blockIdx.x, tid = threadIdx.x;
  const float* lg = logits + (size_t)b * VT;
  float mx = -3.4e38f;
  for (int i = tid; i < VT; i += 256) mx = fmaxf(mx, lg[i]);
  red[tid] = mx; __syncthreads();
  for (int o = 128; o > 0; o >>= 1) { if (tid < o) red[tid] = fmaxf(red[tid], red[tid + o]); __syncthreads(); }
  mx = red[0]; __syncthreads();
  float s = 0.f;
  for (int i = tid; i < VT; i += 256) s += expf(lg[i] - mx);
  red[tid] = s; __syncthreads();
  for (int o = 128; o > 0; o >>= 1) { if (tid < o) red[tid] += red[tid + o]; __syncthreads(); }
  if (tid == 0 && (t + 1) < trg_lens[b]) {
    int tgt = trg_tokens[b * T_ + t + 1];
    atomicAdd(out, lg[tgt] - mx - logf(red[0]));
  }
}

// ---------------------------------------------------------------------------
// Host orchestration
// ---------------------------------------------------------------------------
extern "C" void kernel_launch(void* const* d_in, const int* in_sizes, int n_in,
                              void* d_out, int out_size, void* d_ws, size_t ws_size,
                              hipStream_t stream)
{
  (void)in_sizes; (void)n_in; (void)out_size; (void)ws_size;

  const int*   src_tokens = (const int*)  d_in[0];
  const int*   src_lens   = (const int*)  d_in[1];
  const int*   trg_tokens = (const int*)  d_in[2];
  const int*   trg_lens   = (const int*)  d_in[3];
  const float* src_emb    = (const float*)d_in[4];
  const float* trg_emb    = (const float*)d_in[5];
  const float* enc_Wih0   = (const float*)d_in[6];
  const float* enc_Whh0   = (const float*)d_in[7];
  const float* enc_b0     = (const float*)d_in[8];
  const float* enc_Wih1   = (const float*)d_in[9];
  const float* enc_Whh1   = (const float*)d_in[10];
  const float* enc_b1     = (const float*)d_in[11];
  const float* dec_Wih    = (const float*)d_in[12];
  const float* dec_Whh    = (const float*)d_in[13];
  const float* dec_b      = (const float*)d_in[14];
  const float* hid_W      = (const float*)d_in[15];
  const float* hid_b      = (const float*)d_in[16];
  const float* out_W      = (const float*)d_in[17];
  const float* out_b      = (const float*)d_in[18];
  const float* init_W     = (const float*)d_in[19];
  const float* init_b     = (const float*)d_in[20];

  // ---- bump allocator over workspace ----
  char* ws = (char*)d_ws;
  size_t off = 0;
  auto alloc = [&](size_t bytes) -> void* {
    void* p = ws + off;
    off = (off + bytes + 255) & ~(size_t)255;
    return p;
  };
  const size_t BH = (size_t)B_ * H_;

  _Float16* wih0_16  = (_Float16*)alloc((size_t)G4H * E_  * 2);
  _Float16* whh0_16  = (_Float16*)alloc((size_t)G4H * H_  * 2);
  _Float16* wih1_16  = (_Float16*)alloc((size_t)G4H * H_  * 2);
  _Float16* whh1_16  = (_Float16*)alloc((size_t)G4H * H_  * 2);
  _Float16* dwih_16  = (_Float16*)alloc((size_t)G4H * (E_ + H_) * 2);
  _Float16* dwhh_16  = (_Float16*)alloc((size_t)G4H * H_  * 2);
  _Float16* hidW_16  = (_Float16*)alloc((size_t)H_ * 2 * H_ * 2);
  _Float16* outW_16  = (_Float16*)alloc((size_t)VT * H_  * 2);
  _Float16* initW_16 = (_Float16*)alloc((size_t)H_ * 2 * H_ * 2);

  _Float16* x16        = (_Float16*)alloc((size_t)L_ * B_ * E_ * 2);   // 8 MB
  float*    wx0        = (float*)   alloc((size_t)L_ * B_ * G4H * 4);  // 64 MB
  float*    src_states = (float*)   alloc((size_t)B_ * L_ * H_ * 4);   // 16 MB
  float*    logits     = (float*)   alloc((size_t)B_ * VT * 4);        // 16 MB

  float* h1 = (float*)alloc(BH * 4); float* c1 = (float*)alloc(BH * 4);
  float* h2 = (float*)alloc(BH * 4); float* c2 = (float*)alloc(BH * 4);
  _Float16* h1_16  = (_Float16*)alloc(BH * 2);
  _Float16* nh1_16 = (_Float16*)alloc(BH * 2);
  _Float16* h2_16  = (_Float16*)alloc(BH * 2);

  float* preact = (float*)alloc((size_t)B_ * G4H * 4);
  float* tmp1   = (float*)alloc((size_t)B_ * G4H * 4);

  _Float16* cellcat16 = (_Float16*)alloc(BH * 2 * 2);
  float* h_dec = (float*)alloc(BH * 4);
  float* c_dec = (float*)alloc(BH * 4);
  _Float16* hdec16 = (_Float16*)alloc(BH * 2);
  float* av = (float*)alloc(BH * 4);
  _Float16* av16 = (_Float16*)alloc(BH * 2);
  _Float16* in16 = (_Float16*)alloc(BH * 2 * 2);
  float* attn = (float*)alloc((size_t)B_ * L_ * 4);
  _Float16* cat2_16 = (_Float16*)alloc(BH * 2 * 2);

  auto cvt = [&](const float* s, _Float16* d, int n) {
    f32_to_f16_k<<<(n + 255) / 256, 256, 0, stream>>>(s, d, n);
  };
  auto gemm = [&](const _Float16* A, int lda, const _Float16* W, int ldb,
                  const float* addend, const float* bias,
                  float* C, _Float16* C16, int ldc, int M, int N, int K) {
    dim3 grid(N / 64, M / 64);
    gemm_f16_wmma<<<grid, 128, 0, stream>>>(A, lda, W, ldb, addend, bias, C, C16, ldc, K);
  };

  // ---- init ----
  hipMemsetAsync(d_out, 0, sizeof(float), stream);
  hipMemsetAsync(h1, 0, BH * 4, stream);   hipMemsetAsync(c1, 0, BH * 4, stream);
  hipMemsetAsync(h2, 0, BH * 4, stream);   hipMemsetAsync(c2, 0, BH * 4, stream);
  hipMemsetAsync(h1_16, 0, BH * 2, stream); hipMemsetAsync(h2_16, 0, BH * 2, stream);
  hipMemsetAsync(av, 0, BH * 4, stream);

  // ---- weight conversion (f32 -> f16) ----
  cvt(enc_Wih0, wih0_16, G4H * E_);
  cvt(enc_Whh0, whh0_16, G4H * H_);
  cvt(enc_Wih1, wih1_16, G4H * H_);
  cvt(enc_Whh1, whh1_16, G4H * H_);
  cvt(dec_Wih,  dwih_16, G4H * (E_ + H_));
  cvt(dec_Whh,  dwhh_16, G4H * H_);
  cvt(hid_W,    hidW_16, H_ * 2 * H_);
  cvt(out_W,    outW_16, VT * H_);
  cvt(init_W,   initW_16, H_ * 2 * H_);

  // ---- encoder ----
  gather_src_k<<<(L_ * B_ * E_) / 256, 256, 0, stream>>>(src_tokens, src_emb, x16);
  // batched input-to-hidden for layer 0 over all L steps: (L*B, 4H), bias b0 folded in
  gemm(x16, E_, wih0_16, E_, nullptr, enc_b0, wx0, nullptr, G4H, L_ * B_, G4H, E_);

  const int gateGrid = (int)(BH / 256);
  for (int l = 0; l < L_; ++l) {
    // layer 0: preact = h1 @ Whh0^T + wx0[l]
    gemm(h1_16, H_, whh0_16, H_, wx0 + (size_t)l * B_ * G4H, nullptr,
         preact, nullptr, G4H, B_, G4H, H_);
    lstm_gate_k<<<gateGrid, 256, 0, stream>>>(preact, h1, c1, h1_16, nh1_16,
                                              nullptr, src_lens, l);
    // layer 1: tmp1 = nh1 @ Wih1^T + b1 ; preact = h2 @ Whh1^T + tmp1
    gemm(nh1_16, H_, wih1_16, H_, nullptr, enc_b1, tmp1, nullptr, G4H, B_, G4H, H_);
    gemm(h2_16, H_, whh1_16, H_, tmp1, nullptr, preact, nullptr, G4H, B_, G4H, H_);
    lstm_gate_k<<<gateGrid, 256, 0, stream>>>(preact, h2, c2, h2_16, nullptr,
                                              src_states, src_lens, l);
  }

  // ---- decoder init: c0 = [c1;c2] @ init_W^T + init_b ; h0 = tanh(c0) ----
  cellcat_k<<<(B_ * 2 * H_) / 256, 256, 0, stream>>>(c1, c2, cellcat16);
  gemm(cellcat16, 2 * H_, initW_16, 2 * H_, nullptr, init_b, c_dec, nullptr, H_, B_, H_, 2 * H_);
  tanh_k<<<gateGrid, 256, 0, stream>>>(c_dec, h_dec, hdec16);

  // ---- decoder (teacher forcing, T-1 steps) ----
  for (int t = 0; t < T_ - 1; ++t) {
    dec_input_k<<<B_, 1024, 0, stream>>>(trg_tokens, trg_emb, av, in16, t);
    // preact = [emb;av] @ dWih^T + b  +  h @ dWhh^T
    gemm(in16, E_ + H_, dwih_16, E_ + H_, nullptr, dec_b, tmp1, nullptr, G4H, B_, G4H, E_ + H_);
    gemm(hdec16, H_, dwhh_16, H_, tmp1, nullptr, preact, nullptr, G4H, B_, G4H, H_);
    lstm_gate_k<<<gateGrid, 256, 0, stream>>>(preact, h_dec, c_dec, hdec16, nullptr,
                                              nullptr, nullptr, 0);
    attn_sm_k<<<B_, L_, 0, stream>>>(h_dec, src_states, src_lens, attn);
    ctx_concat_k<<<B_, H_, 0, stream>>>(attn, src_states, h_dec, cat2_16);
    // av = [h;ctx] @ hid_W^T + hid_b  (also produce f16 mirror for next GEMMs)
    gemm(cat2_16, 2 * H_, hidW_16, 2 * H_, nullptr, hid_b, av, av16, H_, B_, H_, 2 * H_);
    // logits = av @ out_W^T + out_b   (the dominant GEMM: 128 x 32000, K=512)
    gemm(av16, H_, outW_16, H_, nullptr, out_b, logits, nullptr, VT, B_, VT, H_);
    nll_step_k<<<B_, 256, 0, stream>>>(logits, trg_tokens, trg_lens, (float*)d_out, t);
  }
}
